// ArticulatoryVQTokenizer_38096359915595
// MI455X (gfx1250) — matrix operand order repaired and checked by reading, not compiled, not run on hardware
//
#include <hip/hip_runtime.h>
#include <hip/hip_bf16.h>
#include <math.h>

typedef _Float16 h16;
typedef __attribute__((ext_vector_type(16))) _Float16 v16h;
typedef __attribute__((ext_vector_type(8)))  float    v8f;

union AFrag { v16h v; h16 e[16]; };
union DFrag { v8f  v; float e[8]; };

#define NTOK_TILE 128
#define HID 128
#define LAT 64
#define KCB 512
#define IN_DIM 14

// ---- LDS layout (bytes); all operand pools are in fragment-contiguous layout:
// ---- halfword address = ((frag*32 + lane)*16 + j), 32B-aligned per lane chunk.
constexpr int OFF_W1  = 0;                      // h16[1*8*512]   enc_w1  (KC=1, NT=8)
constexpr int OFF_W2  = OFF_W1  + 8*512*2;      // h16[4*4*512]   enc_w2  (KC=4, NT=4)
constexpr int OFF_CBT = OFF_W2  + 16*512*2;     // h16[2*32*512]  codebook^T (KC=2, NT=32)
constexpr int OFF_DW1 = OFF_CBT + 64*512*2;     // h16[2*8*512]   dec_w1  (KC=2, NT=8)
constexpr int OFF_DW2 = OFF_DW1 + 16*512*2;     // h16[4*1*512]   dec_w2  (KC=4, NT=1)
constexpr int OFF_B1  = OFF_DW2 + 4*512*2;      // f32[128]
constexpr int OFF_G1  = OFF_B1  + 128*4;
constexpr int OFF_BB1 = OFF_G1  + 128*4;
constexpr int OFF_B2  = OFF_BB1 + 128*4;        // f32[64]
constexpr int OFF_DB1 = OFF_B2  + 64*4;
constexpr int OFF_G2  = OFF_DB1 + 128*4;
constexpr int OFF_BB2 = OFF_G2  + 128*4;
constexpr int OFF_DB2 = OFF_BB2 + 128*4;        // f32[16]
constexpr int OFF_CN  = OFF_DB2 + 16*4;         // f32[512] codebook row norms
constexpr int OFF_XA  = OFF_CN  + 512*4;        // h16[8*512]    x tile, A-frag pool (MT=8, KC=1)
constexpr int OFF_ACT = OFF_XA  + 8*512*2;      // h16[128*128]  activations, row-major (LN)
constexpr int OFF_ZF  = OFF_ACT + 128*128*2;    // f32[128*64]   z row-major (commit loss)
constexpr int OFF_ZA  = OFF_ZF  + 128*64*4;     // h16[16*512]   z A-frag pool (MT=8, KC=2)
constexpr int OFF_QH  = OFF_ZA  + 16*512*2;     // h16[16*512]   q A-frag pool (MT=8, KC=2)
constexpr int OFF_IDX = OFF_QH  + 16*512*2;     // i32[128]
constexpr int OFF_RED = OFF_IDX + 128*4;        // f32[256]
constexpr int SMEM_BYTES = OFF_RED + 256*4;     // ~219 KB (fits 320 KB WGP LDS)

// ---- WMMA fragment maps (CDNA5 ISA 7.12.2) ----
// A 16-bit 16x32: lane half h, slot j -> k = j + 8h (j<8) | j + 8 + 8h (j>=8)
// B 16-bit 32x16: lane half h, slot j -> same k as A (k = j + 8h + (j & 8))
__device__ __forceinline__ int kmapA(int lane, int j) {
  int h = (lane >> 4) & 1;
  return j + (h << 3) + (j & 8);
}
// inverse of kmapA: k in [0,32) -> (lane half h, slot j)
__device__ __forceinline__ void invKA(int k, int& h, int& j) {
  h = (k >> 3) & 1;
  j = (k & 7) + (((k >> 4) & 1) << 3);
}

// fragment load: one aligned 32B read (2x ds_load_b128), no packing moves
__device__ __forceinline__ v16h loadFrag(const h16* pool, int f, int lane) {
  return *(const v16h*)(pool + (size_t)((f*32 + lane) << 4));
}
// scalar A gather (only used for row-major `act`)
__device__ __forceinline__ v16h loadA(const h16* A, int ldk, int m0, int kbase, int lane) {
  AFrag f; const h16* p = A + (m0 + (lane & 15))*ldk + kbase;
#pragma unroll
  for (int j = 0; j < 16; ++j) f.e[j] = p[kmapA(lane, j)];
  return f.v;
}
__device__ __forceinline__ v8f wmma16(v16h a, v16h b, v8f c) {
  return __builtin_amdgcn_wmma_f32_16x16x32_f16(false, a, false, b, (short)0, c, false, false);
}
__device__ __forceinline__ void storeD_h16(h16* dst, int ld, int m0, int n0, int lane,
                                           const DFrag& d, const float* bias) {
  int n = n0 + (lane & 15), hf = (lane >> 4) & 1;
  float b = bias[n];
#pragma unroll
  for (int r = 0; r < 8; ++r) dst[(m0 + r + 8*hf)*ld + n] = (h16)(d.e[r] + b);
}

// per-wave LayerNorm + exact GELU over the wave's 16x128 rows (in-place, f16)
__device__ __forceinline__ void ln_gelu_wave(h16* act, int m0, int lane,
                                             const float* g, const float* bb) {
  int row = m0 + (lane >> 1);
  int seg = lane & 1;
  h16* rp = act + row*HID + seg*64;
  float s = 0.f, s2 = 0.f;
#pragma unroll 8
  for (int f = 0; f < 64; ++f) { float v = (float)rp[f]; s += v; s2 += v*v; }
  s  += __shfl_xor(s, 1, 32);
  s2 += __shfl_xor(s2, 1, 32);
  float mu  = s  * (1.f/128.f);
  float var = s2 * (1.f/128.f) - mu*mu;
  float rs  = rsqrtf(var + 1e-5f);
  const float* gg = g + seg*64; const float* bv = bb + seg*64;
#pragma unroll 4
  for (int f = 0; f < 64; ++f) {
    float v = ((float)rp[f] - mu) * rs * gg[f] + bv[f];
    float ge = 0.5f * v * (1.f + erff(v * 0.70710678118f));
    rp[f] = (h16)ge;
  }
}

// swizzle a B matrix into fragment pool: dst[e], e = ((kc*NT+nt)*32+lane)*16+j
// holds B[kc*32 + j + 8*(lane>>4) + (j&8)][nt*16 + (lane&15)]
#define FILL_B(dst, KC, NT, EXPR)                                        \
  for (int e = tid; e < (KC)*(NT)*512; e += 256) {                       \
    int f_ = e >> 9, ln_ = (e >> 4) & 31, j = e & 15;                    \
    int kc = f_ / (NT), nt = f_ % (NT);                                  \
    int k = kc*32 + j + ((ln_ >> 4) << 3) + (j & 8);                     \
    int n = nt*16 + (ln_ & 15);                                          \
    (dst)[e] = (h16)(EXPR);                                              \
  }

__global__ void __launch_bounds__(256)
vq_fused(const float* __restrict__ x,
         const float* __restrict__ enc_w1, const float* __restrict__ enc_b1,
         const float* __restrict__ ln1_g,  const float* __restrict__ ln1_b,
         const float* __restrict__ enc_w2, const float* __restrict__ enc_b2,
         const float* __restrict__ codebook,
         const float* __restrict__ dec_w1, const float* __restrict__ dec_b1,
         const float* __restrict__ ln2_g,  const float* __restrict__ ln2_b,
         const float* __restrict__ dec_w2, const float* __restrict__ dec_b2,
         float* __restrict__ out_rec, float* __restrict__ out_idx,
         unsigned* __restrict__ hist, float* __restrict__ partial) {
  extern __shared__ __align__(32) char smem[];
  h16*   ls_w1  = (h16*)(smem + OFF_W1);
  h16*   ls_w2  = (h16*)(smem + OFF_W2);
  h16*   ls_cbt = (h16*)(smem + OFF_CBT);
  h16*   ls_dw1 = (h16*)(smem + OFF_DW1);
  h16*   ls_dw2 = (h16*)(smem + OFF_DW2);
  float* ls_b1  = (float*)(smem + OFF_B1);
  float* ls_g1  = (float*)(smem + OFF_G1);
  float* ls_bb1 = (float*)(smem + OFF_BB1);
  float* ls_b2  = (float*)(smem + OFF_B2);
  float* ls_db1 = (float*)(smem + OFF_DB1);
  float* ls_g2  = (float*)(smem + OFF_G2);
  float* ls_bb2 = (float*)(smem + OFF_BB2);
  float* ls_db2 = (float*)(smem + OFF_DB2);
  float* ls_cn  = (float*)(smem + OFF_CN);
  h16*   ls_xa  = (h16*)(smem + OFF_XA);
  h16*   ls_act = (h16*)(smem + OFF_ACT);
  float* ls_zf  = (float*)(smem + OFF_ZF);
  h16*   ls_za  = (h16*)(smem + OFF_ZA);
  h16*   ls_qh  = (h16*)(smem + OFF_QH);
  int*   ls_idx = (int*)(smem + OFF_IDX);
  float* ls_red = (float*)(smem + OFF_RED);

  const int tid  = threadIdx.x;
  const int lane = tid & 31;
  const int wave = tid >> 5;
  const int m0   = wave * 16;
  const int hf   = (lane >> 4) & 1;
  const int tok0 = blockIdx.x * NTOK_TILE;

  // ---- stage 0: weights -> LDS, pre-swizzled into WMMA fragment layout ----
  FILL_B(ls_w1,  1, 8,  (k < IN_DIM) ? enc_w1[k*HID + n] : 0.f)
  FILL_B(ls_w2,  4, 4,  enc_w2[k*LAT + n])
  FILL_B(ls_cbt, 2, 32, codebook[n*LAT + k])
  FILL_B(ls_dw1, 2, 8,  dec_w1[k*HID + n])
  FILL_B(ls_dw2, 4, 1,  (n < IN_DIM) ? dec_w2[k*IN_DIM + n] : 0.f)
  for (int i = tid; i < 128; i += 256) {
    ls_b1[i] = enc_b1[i]; ls_g1[i] = ln1_g[i]; ls_bb1[i] = ln1_b[i];
    ls_db1[i] = dec_b1[i]; ls_g2[i] = ln2_g[i]; ls_bb2[i] = ln2_b[i];
  }
  for (int i = tid; i < 64; i += 256) ls_b2[i] = enc_b2[i];
  for (int i = tid; i < 16; i += 256) ls_db2[i] = (i < IN_DIM) ? dec_b2[i] : 0.f;
  for (int c = tid; c < 512; c += 256) {
    float s = 0.f;
    for (int k = 0; k < LAT; ++k) { float v = codebook[c*LAT + k]; s += v*v; }
    ls_cn[c] = s;
  }
  __syncthreads();

  // ---- x tile -> A-fragment pool (MT=8, KC=1; K padded 14->32) ----
  for (int e = tid; e < 8*512; e += 256) {
    int f_ = e >> 9, ln_ = (e >> 4) & 31, j = e & 15;
    int row = f_*16 + (ln_ & 15);
    int k = j + ((ln_ >> 4) << 3) + (j & 8);   // kmapA
    ls_xa[e] = (h16)((k < IN_DIM) ? x[(size_t)(tok0 + row)*IN_DIM + k] : 0.f);
  }
  __syncthreads();

  // ---- enc1: (128x32) @ (32x128) -> act (row-major, for LN) ----
  {
    v16h a = loadFrag(ls_xa, wave, lane);
#pragma unroll
    for (int nt = 0; nt < 8; ++nt) {
      DFrag d;
#pragma unroll
      for (int r = 0; r < 8; ++r) d.e[r] = 0.f;
      d.v = wmma16(a, loadFrag(ls_w1, nt, lane), d.v);
      storeD_h16(ls_act, HID, m0, nt*16, lane, d, ls_b1);
    }
  }
  __syncthreads();
  ln_gelu_wave(ls_act, m0, lane, ls_g1, ls_bb1);
  __syncthreads();

  // ---- enc2: (128x128) @ (128x64) -> z (f32 row-major + f16 A-frag pool) ----
  {
    v16h a0 = loadA(ls_act, HID, m0, 0,  lane);
    v16h a1 = loadA(ls_act, HID, m0, 32, lane);
    v16h a2 = loadA(ls_act, HID, m0, 64, lane);
    v16h a3 = loadA(ls_act, HID, m0, 96, lane);
#pragma unroll
    for (int nt = 0; nt < 4; ++nt) {
      DFrag d;
#pragma unroll
      for (int r = 0; r < 8; ++r) d.e[r] = 0.f;
      d.v = wmma16(a0, loadFrag(ls_w2, 0*4 + nt, lane), d.v);
      d.v = wmma16(a1, loadFrag(ls_w2, 1*4 + nt, lane), d.v);
      d.v = wmma16(a2, loadFrag(ls_w2, 2*4 + nt, lane), d.v);
      d.v = wmma16(a3, loadFrag(ls_w2, 3*4 + nt, lane), d.v);
      int n = nt*16 + (lane & 15);
      float bb = ls_b2[n];
      int hh, jj; invKA(n & 31, hh, jj);
      int fz = wave*2 + (n >> 5);
#pragma unroll
      for (int r = 0; r < 8; ++r) {
        int row = m0 + r + 8*hf;
        float zv = d.e[r] + bb;
        ls_zf[row*LAT + n] = zv;
        ls_za[((fz*32 + (row & 15) + 16*hh) << 4) + jj] = (h16)zv;
      }
    }
  }
  __syncthreads();

  // ---- VQ: argmin_c ( ||c||^2 - 2 z.c )  via z @ cb^T  (128x64 @ 64x512) ----
  float best[8]; int bidx[8];
#pragma unroll
  for (int r = 0; r < 8; ++r) { best[r] = 3.4e38f; bidx[r] = 0; }
  {
    v16h a0 = loadFrag(ls_za, wave*2 + 0, lane);
    v16h a1 = loadFrag(ls_za, wave*2 + 1, lane);
    for (int ct = 0; ct < 32; ++ct) {
      DFrag d;
#pragma unroll
      for (int r = 0; r < 8; ++r) d.e[r] = 0.f;
      d.v = wmma16(a0, loadFrag(ls_cbt, ct,      lane), d.v);
      d.v = wmma16(a1, loadFrag(ls_cbt, 32 + ct, lane), d.v);
      int c = ct*16 + (lane & 15);
      float cn = ls_cn[c];
#pragma unroll
      for (int r = 0; r < 8; ++r) {
        float score = cn - 2.f*d.e[r];
        if (score < best[r]) { best[r] = score; bidx[r] = c; }
      }
    }
  }
#pragma unroll
  for (int r = 0; r < 8; ++r) {
    float bv = best[r]; int bi = bidx[r];
#pragma unroll
    for (int off = 8; off > 0; off >>= 1) {
      float ov = __shfl_xor(bv, off, 16);
      int   oi = __shfl_xor(bi, off, 16);
      if (ov < bv || (ov == bv && oi < bi)) { bv = ov; bi = oi; }
    }
    if ((lane & 15) == 0) {
      int row = m0 + r + 8*hf;
      ls_idx[row] = bi;
      out_idx[tok0 + row] = (float)bi;
      atomicAdd(&hist[bi], 1u);
    }
  }
  __syncthreads();

  // ---- q lookup (f32 from L2-resident codebook) + commitment-loss partial ----
  float csum = 0.f;
  for (int i = tid; i < NTOK_TILE*LAT; i += 256) {
    int row = i >> 6, k = i & 63;
    int bi = ls_idx[row];
    float qv = codebook[(size_t)bi*LAT + k];
    int hh, jj; invKA(k & 31, hh, jj);
    int fq = (row >> 4)*2 + (k >> 5);
    ls_qh[((fq*32 + (row & 15) + 16*hh) << 4) + jj] = (h16)qv;
    float dd = qv - ls_zf[i];
    csum += dd*dd;
  }
  ls_red[tid] = csum;
  __syncthreads();
  for (int s = 128; s > 0; s >>= 1) { if (tid < s) ls_red[tid] += ls_red[tid + s]; __syncthreads(); }
  if (tid == 0) partial[blockIdx.x] = ls_red[0];

  // ---- dec1: (128x64) @ (64x128) -> act ----
  {
    v16h a0 = loadFrag(ls_qh, wave*2 + 0, lane);
    v16h a1 = loadFrag(ls_qh, wave*2 + 1, lane);
#pragma unroll
    for (int nt = 0; nt < 8; ++nt) {
      DFrag d;
#pragma unroll
      for (int r = 0; r < 8; ++r) d.e[r] = 0.f;
      d.v = wmma16(a0, loadFrag(ls_dw1, nt,     lane), d.v);
      d.v = wmma16(a1, loadFrag(ls_dw1, 8 + nt, lane), d.v);
      storeD_h16(ls_act, HID, m0, nt*16, lane, d, ls_db1);
    }
  }
  __syncthreads();
  ln_gelu_wave(ls_act, m0, lane, ls_g2, ls_bb2);
  __syncthreads();

  // ---- dec2: (128x128) @ (128x16) -> reconstruction ----
  {
    DFrag d;
#pragma unroll
    for (int r = 0; r < 8; ++r) d.e[r] = 0.f;
#pragma unroll
    for (int kc = 0; kc < 4; ++kc) {
      v16h a = loadA(ls_act, HID, m0, kc*32, lane);
      d.v = wmma16(a, loadFrag(ls_dw2, kc, lane), d.v);
    }
    int n = lane & 15;
    if (n < IN_DIM) {
      float bb = ls_db2[n];
#pragma unroll
      for (int r = 0; r < 8; ++r) {
        int row = m0 + r + 8*hf;
        out_rec[(size_t)(tok0 + row)*IN_DIM + n] = d.e[r] + bb;
      }
    }
  }
}

__global__ void __launch_bounds__(512)
vq_finalize(const unsigned* __restrict__ hist, const float* __restrict__ partial,
            int nblocks, int ntok, float* __restrict__ out_sc) {
  __shared__ float sh[512];
  int t = threadIdx.x;
  // perplexity from histogram
  float p = (float)hist[t] / (float)ntok;
  sh[t] = p * logf(p + 1e-10f);
  __syncthreads();
  for (int s = 256; s > 0; s >>= 1) { if (t < s) sh[t] += sh[t + s]; __syncthreads(); }
  float neg_ent = sh[0];
  __syncthreads();
  // commitment loss from deterministic fixed-order partial sum
  float c = 0.f;
  for (int i = t; i < nblocks; i += 512) c += partial[i];
  sh[t] = c;
  __syncthreads();
  for (int s = 256; s > 0; s >>= 1) { if (t < s) sh[t] += sh[t + s]; __syncthreads(); }
  if (t == 0) {
    out_sc[0] = 0.25f * sh[0] / ((float)ntok * (float)LAT);
    out_sc[1] = expf(-neg_ent);
  }
}

extern "C" void kernel_launch(void* const* d_in, const int* in_sizes, int n_in,
                              void* d_out, int out_size, void* d_ws, size_t ws_size,
                              hipStream_t stream) {
  const float* x       = (const float*)d_in[0];
  const float* enc_w1  = (const float*)d_in[1];
  const float* enc_b1  = (const float*)d_in[2];
  const float* ln1_g   = (const float*)d_in[3];
  const float* ln1_b   = (const float*)d_in[4];
  const float* enc_w2  = (const float*)d_in[5];
  const float* enc_b2  = (const float*)d_in[6];
  const float* codebook= (const float*)d_in[7];
  const float* dec_w1  = (const float*)d_in[8];
  const float* dec_b1  = (const float*)d_in[9];
  const float* ln2_g   = (const float*)d_in[10];
  const float* ln2_b   = (const float*)d_in[11];
  const float* dec_w2  = (const float*)d_in[12];
  const float* dec_b2  = (const float*)d_in[13];

  float* out = (float*)d_out;
  const int n_tok  = in_sizes[0] / IN_DIM;   // 131072
  const int blocks = n_tok / NTOK_TILE;      // 1024

  unsigned* hist   = (unsigned*)d_ws;
  float*    partial= (float*)((char*)d_ws + 512*sizeof(unsigned));
  hipMemsetAsync(d_ws, 0, 512*sizeof(unsigned) + (size_t)blocks*sizeof(float), stream);

  float* out_rec = out;
  float* out_idx = out + (size_t)n_tok*IN_DIM;
  float* out_sc  = out_idx + n_tok;

  hipFuncSetAttribute((const void*)vq_fused,
                      hipFuncAttributeMaxDynamicSharedMemorySize, SMEM_BYTES);

  vq_fused<<<blocks, 256, SMEM_BYTES, stream>>>(
      x, enc_w1, enc_b1, ln1_g, ln1_b, enc_w2, enc_b2, codebook,
      dec_w1, dec_b1, ln2_g, ln2_b, dec_w2, dec_b2,
      out_rec, out_idx, hist, partial);

  vq_finalize<<<1, 512, 0, stream>>>(hist, partial, blocks, n_tok, out_sc);
}
